// TransformerDecoder_52261162058450
// MI455X (gfx1250) — matrix operand order repaired and checked by reading, not compiled, not run on hardware
//
#include <hip/hip_runtime.h>
#include <hip/hip_bf16.h>
#include <stdint.h>

typedef __bf16 bf16;
typedef __attribute__((ext_vector_type(8)))  __bf16 v8bf;
typedef __attribute__((ext_vector_type(16))) __bf16 v16bf;
typedef __attribute__((ext_vector_type(8)))  float  v8f;

namespace {

constexpr int Bn = 16, Sq = 512, Hd = 1024, Ln = 4, Vc = 10000;
constexpr float kNegInf = -10000000.0f;

__device__ __forceinline__ v8bf pack8(float4 a, float4 b) {
  v8bf r;
  r[0] = (bf16)a.x; r[1] = (bf16)a.y; r[2] = (bf16)a.z; r[3] = (bf16)a.w;
  r[4] = (bf16)b.x; r[5] = (bf16)b.y; r[6] = (bf16)b.z; r[7] = (bf16)b.w;
  return r;
}

__device__ __forceinline__ unsigned lds_addr32(const void* p) {
  // Generic pointers to LDS carry the wave-relative LDS byte address in the
  // low 32 bits (ISA 10.2, LDS aperture).
  return (unsigned)(uintptr_t)p;
}

// Async bf16 tile copy: 16 bytes global -> LDS, tracked by ASYNCcnt.
__device__ __forceinline__ void async_copy_b128(const bf16* g, void* lds) {
  asm volatile("global_load_async_to_lds_b128 %0, %1, off"
               :: "v"(lds_addr32(lds)), "v"(g)
               : "memory");
}

// ---------------------------------------------------------------------------
// f32 -> bf16 mirror (n must be a multiple of 8; true for all arrays here).
// ---------------------------------------------------------------------------
__global__ void cvt_bf16_kernel(const float* __restrict__ in,
                                bf16* __restrict__ out, long n) {
  long i = ((long)blockIdx.x * blockDim.x + threadIdx.x) * 8;
  if (i >= n) return;
  const float4* g = (const float4*)(in + i);
  *(v8bf*)(out + i) = pack8(g[0], g[1]);
}

// ---------------------------------------------------------------------------
// x[b,s,:] = emb[tok[b,s],:] + PE(s,:)   (f32 master + bf16 mirror)
// ---------------------------------------------------------------------------
__global__ void embed_pe_kernel(const int* __restrict__ tok,
                                const float* __restrict__ emb,
                                float* __restrict__ x,
                                bf16* __restrict__ xb) {
  long i = (long)blockIdx.x * blockDim.x + threadIdx.x;
  const long total = (long)Bn * Sq * Hd;
  if (i >= total) return;
  int h = (int)(i % Hd);
  long bs = i / Hd;
  int s = (int)(bs % Sq);
  int t = tok[bs];
  int j = h >> 1;
  float ang = (float)s * __powf(10000.0f, -(2.0f * (float)j) / (float)Hd);
  float pe = (h & 1) ? __cosf(ang) : __sinf(ang);
  float v = emb[(long)t * Hd + h] + pe;
  x[i] = v;
  xb[i] = (bf16)v;
}

// ---------------------------------------------------------------------------
// Tiled bf16 WMMA GEMM over bf16-resident operands.
// Double-buffered async global->LDS staging: tile t+1's DMA is issued before
// waiting on tile t (s_wait_asynccnt <= pend; async loads complete in order),
// so the copies for the next tile are in flight during this tile's WMMAs.
//   C[z][m,n] = epi( scale * sum_k A[z][m,k] * B'[z][k,n] )
//   B' = B (row-major KxN) if !TB, else B^T of row-major NxK (Q·K^T path).
//   epi: +bias[n]; causal (n>=m -> +NEG_INF); ReLU; +residual[m,n];
//        optional f32 store (C) and/or bf16 store (Cb).
// Block tile 128x128, 256 thr = 8 wave32s, wave tile 32x64 = 8 WMMAs/K-step.
// ASSUMES (true at every call site): M % 128 == 0, K % 32 == 0; only N may
// have a ragged edge (Vc = 10000), handled by guarded staging/epilogue.
// ---------------------------------------------------------------------------
template <bool TB>
__global__ __launch_bounds__(256) void gemm_bf16_wmma(
    const bf16* __restrict__ A, const bf16* __restrict__ Bm,
    const float* __restrict__ bias, const float* __restrict__ res,
    float* __restrict__ C, bf16* __restrict__ Cb,
    int M, int N, int K, long sA, long sB, long sC,
    float scale, int causal, int relu) {
  __shared__ bf16 As[2][128][40];  // 128x32 (MxK), row stride 80B (16B mult)
  __shared__ bf16 Bs[2][32][136];  // 32x128 (KxN), row stride 272B (16B mult)

  const int z = blockIdx.z;
  A += (long)z * sA;
  Bm += (long)z * sB;

  const int tid = threadIdx.x;
  const int lane = tid & 31;
  const int wave = tid >> 5;
  const int wm = (wave >> 1) * 32;  // wave M offset in block tile
  const int wn = (wave & 1) * 64;   // wave N offset in block tile
  const int tileM = blockIdx.y * 128;
  const int tileN = blockIdx.x * 128;
  const bool edgeN = !TB && (tileN + 128 > N);

  // Stage one 128x32 A tile + 32x128 B tile into LDS buffer `buf`.
  auto stage = [&](int k0, int buf) {
    // A: pure bf16 copy via async DMA (2 x b128 per thread, 8 KB total).
    #pragma unroll
    for (int c = 0; c < 2; ++c) {
      const int ch = tid + c * 256;        // 512 chunks of 8 bf16
      const int row = ch >> 2, seg = ch & 3;
      async_copy_b128(A + (long)(tileM + row) * K + k0 + seg * 8,
                      &As[buf][row][seg * 8]);
    }
    if constexpr (TB) {
      // B'(k,n) = Bm[n*K + k]: vector-read along K, transpose into LDS.
      // Only used for Q·K^T (N = 512, tile-aligned): no guards.
      const int nrow = tid >> 1, half = tid & 1;
      const v8bf* g =
          (const v8bf*)(Bm + (long)(tileN + nrow) * K + k0 + half * 16);
      v8bf lo = g[0], hi = g[1];
      #pragma unroll
      for (int e = 0; e < 8; ++e) {
        Bs[buf][half * 16 + e][nrow] = lo[e];
        Bs[buf][half * 16 + 8 + e][nrow] = hi[e];
      }
    } else if (!edgeN) {
      #pragma unroll
      for (int c = 0; c < 2; ++c) {
        const int ch = tid + c * 256;      // 512 chunks of 8 bf16
        const int row = ch >> 4, seg = ch & 15;
        async_copy_b128(Bm + (long)(k0 + row) * N + tileN + seg * 8,
                        &Bs[buf][row][seg * 8]);
      }
    } else {
      // Ragged N edge (final vocab projection only): guarded scalar staging.
      const int row = tid >> 3, seg = tid & 7;
      const int gn0 = tileN + seg * 16;
      const bf16* g = Bm + (long)(k0 + row) * N + gn0;
      #pragma unroll
      for (int e = 0; e < 16; ++e) {
        bf16 v = (bf16)0.0f;
        if (gn0 + e < N) v = g[e];
        Bs[buf][row][seg * 16 + e] = v;
      }
    }
  };

  v8f acc[2][4] = {};

  const int nT = K >> 5;  // K-steps of 32
  stage(0, 0);
  for (int t = 0; t < nT; ++t) {
    const int cur = t & 1;
    // Issue next tile's DMA into the alternate buffer before waiting.
    int pend = 0;
    if (t + 1 < nT) {
      stage((t + 1) << 5, cur ^ 1);
      pend = TB ? 2 : (edgeN ? 2 : 4);  // async instructions just issued
    }
    // Async loads complete in order: waiting down to `pend` outstanding
    // guarantees tile t's copies have landed while t+1's stay in flight.
    if (pend == 4)      asm volatile("s_wait_asynccnt 0x4" ::: "memory");
    else if (pend == 2) asm volatile("s_wait_asynccnt 0x2" ::: "memory");
    else                asm volatile("s_wait_asynccnt 0x0" ::: "memory");
    __syncthreads();

    // ---- fragments from LDS (all b128 reads) ----
    // A 16-bit 16x32 layout: lane m = lane&15; 8 contiguous K at
    // kb=(lane>=16)*8 and 16+kb  -> two v8bf loads per fragment.
    const int kb = (lane >> 4) * 8;
    v16bf af[2];
    #pragma unroll
    for (int f = 0; f < 2; ++f) {
      const int mrow = wm + f * 16 + (lane & 15);
      v8bf lo = *(const v8bf*)&As[cur][mrow][kb];
      v8bf hi = *(const v8bf*)&As[cur][mrow][16 + kb];
      af[f] = __builtin_shufflevector(lo, hi, 0, 1, 2, 3, 4, 5, 6, 7, 8, 9, 10,
                                      11, 12, 13, 14, 15);
    }
    // B 16-bit 32x16 layout: lane&15 = N column, elements = K at kb..kb+7
    // and 16+kb..16+kb+7 (same K order as A).
    #pragma unroll
    for (int j = 0; j < 4; ++j) {
      const int nb = wn + j * 16;
      const int bn = nb + (lane & 15);
      v16bf bfj;
      #pragma unroll
      for (int e = 0; e < 8; ++e) {
        bfj[e] = Bs[cur][kb + e][bn];
        bfj[8 + e] = Bs[cur][16 + kb + e][bn];
      }
      #pragma unroll
      for (int f = 0; f < 2; ++f) {
        acc[f][j] = __builtin_amdgcn_wmma_f32_16x16x32_bf16(
            false, af[f], false, bfj, (short)0, acc[f][j], false, false);
      }
    }
    // Keep any wave from re-staging a buffer others are still reading.
    __syncthreads();
  }

  // ---- epilogue. C/D layout: n = lane&15, m = r + 8*(lane>=16) ----
  const float* resz = res ? res + (long)z * sC : nullptr;
  float* cz = C ? C + (long)z * sC : nullptr;
  bf16* cbz = Cb ? Cb + (long)z * sC : nullptr;
  #pragma unroll
  for (int f = 0; f < 2; ++f) {
    const int mbase = tileM + wm + f * 16 + ((lane >> 4) * 8);
    #pragma unroll
    for (int j = 0; j < 4; ++j) {
      const int n = tileN + wn + j * 16 + (lane & 15);
      if (n >= N) continue;
      const float bv = bias ? bias[n] : 0.0f;
      #pragma unroll
      for (int r = 0; r < 8; ++r) {
        const int m = mbase + r;
        float v = acc[f][j][r] * scale + bv;
        if (causal && n >= m) v += kNegInf;  // triu incl. diagonal
        if (relu) v = fmaxf(v, 0.0f);
        if (resz) v += resz[(long)m * N + n];
        const long idx = (long)m * N + n;
        if (cz) cz[idx] = v;
        if (cbz) cbz[idx] = (bf16)v;
      }
    }
  }
}

// ---------------------------------------------------------------------------
// softmax over the QUERY axis (axis=1 of [B, S, Sk]); one wave per (b, k)
// column, strided by Sk. Dual output: f32 (model output) + bf16 (GEMM feed).
// ---------------------------------------------------------------------------
__global__ void softmax_q_kernel(const float* __restrict__ sc,
                                 float* __restrict__ w,
                                 bf16* __restrict__ wb, int Sk) {
  const int col = blockIdx.x * 8 + (threadIdx.x >> 5);
  const int lane = threadIdx.x & 31;
  if (col >= Bn * Sk) return;
  const int b = col / Sk;
  const int k = col % Sk;
  const long base = ((long)b * Sq) * Sk + k;

  float vals[16];
  float mx = -3.4e38f;
  #pragma unroll
  for (int i = 0; i < 16; ++i) {
    vals[i] = sc[base + (long)(lane + i * 32) * Sk];
    mx = fmaxf(mx, vals[i]);
  }
  #pragma unroll
  for (int off = 16; off; off >>= 1) mx = fmaxf(mx, __shfl_xor(mx, off, 32));

  float sum = 0.0f;
  #pragma unroll
  for (int i = 0; i < 16; ++i) {
    vals[i] = __expf(vals[i] - mx);
    sum += vals[i];
  }
  #pragma unroll
  for (int off = 16; off; off >>= 1) sum += __shfl_xor(sum, off, 32);

  const float inv = 1.0f / sum;
  #pragma unroll
  for (int i = 0; i < 16; ++i) {
    const long idx = base + (long)(lane + i * 32) * Sk;
    const float v = vals[i] * inv;
    w[idx] = v;
    wb[idx] = (bf16)v;
  }
}

void launch_gemm(hipStream_t st, const bf16* A, const bf16* Bm,
                 const float* bias, const float* res, float* C, bf16* Cb,
                 int M, int N, int K, int batch,
                 long sA, long sB, long sC,
                 float scale, bool tb, bool causal, bool relu) {
  dim3 grid((N + 127) / 128, (M + 127) / 128, batch);
  if (tb)
    gemm_bf16_wmma<true><<<grid, 256, 0, st>>>(
        A, Bm, bias, res, C, Cb, M, N, K, sA, sB, sC, scale, causal, relu);
  else
    gemm_bf16_wmma<false><<<grid, 256, 0, st>>>(
        A, Bm, bias, res, C, Cb, M, N, K, sA, sB, sC, scale, causal, relu);
}

void launch_cvt(hipStream_t st, const float* in, bf16* out, long n) {
  long blocks = (n / 8 + 255) / 256;
  cvt_bf16_kernel<<<(unsigned)blocks, 256, 0, st>>>(in, out, n);
}

}  // namespace

extern "C" void kernel_launch(void* const* d_in, const int* in_sizes, int n_in,
                              void* d_out, int out_size, void* d_ws,
                              size_t ws_size, hipStream_t stream) {
  (void)in_sizes; (void)n_in; (void)out_size; (void)ws_size;

  const int*   tok   = (const int*)  d_in[0];
  const float* ann   = (const float*)d_in[1];
  // d_in[2] hidden_init, d_in[3] cell_init: unused by the reference.
  const float* emb   = (const float*)d_in[4];
  const float* Wq_s  = (const float*)d_in[5];
  const float* bq_s  = (const float*)d_in[6];
  const float* Wk_s  = (const float*)d_in[7];
  const float* bk_s  = (const float*)d_in[8];
  const float* Wv_s  = (const float*)d_in[9];
  const float* bv_s  = (const float*)d_in[10];
  const float* Wq_e  = (const float*)d_in[11];
  const float* bq_e  = (const float*)d_in[12];
  const float* Wk_e  = (const float*)d_in[13];
  const float* bk_e  = (const float*)d_in[14];
  const float* Wv_e  = (const float*)d_in[15];
  const float* bv_e  = (const float*)d_in[16];
  const float* Wm    = (const float*)d_in[17];
  const float* bm    = (const float*)d_in[18];
  const float* Wout  = (const float*)d_in[19];
  const float* bout  = (const float*)d_in[20];

  float* out = (float*)d_out;
  const long MQ = (long)Bn * Sq;           // 8192 token rows
  const long nX = MQ * Hd;                 // 8388608
  const long nSc = (long)Bn * Sq * Sq;     // 4194304
  const long nWL = (long)Ln * Hd * Hd;     // per weight stack

  // Output regions: [output | enc_ws | self_ws]
  float* out_logits = out;
  float* enc_w_base = out + (long)Bn * Sq * Vc;
  float* self_w_base = enc_w_base + (long)Ln * Bn * Sq * Sq;

  // ---- workspace bump allocator ----
  char* p = (char*)d_ws;
  auto alloc_f = [&](long n) { float* r = (float*)p; p += n * 4; return r; };
  auto alloc_b = [&](long n) { bf16* r = (bf16*)p; p += n * 2; return r; };

  float* x0 = alloc_f(nX);        // f32 residual master (ping)
  float* x1 = alloc_f(nX);        // f32 residual master (pong)
  float* sc = alloc_f(nSc);       // attention scores
  bf16* xb0 = alloc_b(nX);        // bf16 mirror (ping)
  bf16* xb1 = alloc_b(nX);        // bf16 mirror (pong)
  bf16* annb = alloc_b(nX);       // bf16 annotations
  bf16* qb = alloc_b(nX);
  bf16* kb = alloc_b(nX);
  bf16* vb = alloc_b(nX);
  bf16* wb = alloc_b(nSc);        // bf16 softmax weights
  bf16* Wq_sb = alloc_b(nWL);
  bf16* Wk_sb = alloc_b(nWL);
  bf16* Wv_sb = alloc_b(nWL);
  bf16* Wq_eb = alloc_b(nWL);
  bf16* Wk_eb = alloc_b(nWL);
  bf16* Wv_eb = alloc_b(nWL);
  bf16* Wmb = alloc_b(nWL);
  bf16* Woutb = alloc_b((long)Hd * Vc);

  // ---- one-time conversions (weights + annotations) ----
  launch_cvt(stream, Wq_s, Wq_sb, nWL);
  launch_cvt(stream, Wk_s, Wk_sb, nWL);
  launch_cvt(stream, Wv_s, Wv_sb, nWL);
  launch_cvt(stream, Wq_e, Wq_eb, nWL);
  launch_cvt(stream, Wk_e, Wk_eb, nWL);
  launch_cvt(stream, Wv_e, Wv_eb, nWL);
  launch_cvt(stream, Wm, Wmb, nWL);
  launch_cvt(stream, Wout, Woutb, (long)Hd * Vc);
  launch_cvt(stream, ann, annb, nX);

  embed_pe_kernel<<<(unsigned)((nX + 255) / 256), 256, 0, stream>>>(tok, emb,
                                                                    x0, xb0);

  float* xc = x0;  float* xa = x1;
  bf16* xbc = xb0; bf16* xba = xb1;
  const float scal = 0.03125f;  // 1/sqrt(1024)
  const long sBH = (long)Sq * Hd;
  const long sBS = (long)Sq * Sq;

  for (int l = 0; l < Ln; ++l) {
    const long wOff = (long)l * Hd * Hd;
    const long bOff = (long)l * Hd;

    // ---- self-attention (causal, softmax over query axis) ----
    launch_gemm(stream, xbc, Wq_sb + wOff, bq_s + bOff, nullptr, nullptr, qb,
                (int)MQ, Hd, Hd, 1, 0, 0, 0, 1.0f, false, false, false);
    launch_gemm(stream, xbc, Wk_sb + wOff, bk_s + bOff, nullptr, nullptr, kb,
                (int)MQ, Hd, Hd, 1, 0, 0, 0, 1.0f, false, false, false);
    launch_gemm(stream, xbc, Wv_sb + wOff, bv_s + bOff, nullptr, nullptr, vb,
                (int)MQ, Hd, Hd, 1, 0, 0, 0, 1.0f, false, false, false);
    // scores[b] = scale * q[b] @ k[b]^T + causal mask
    launch_gemm(stream, qb, kb, nullptr, nullptr, sc, nullptr,
                Sq, Sq, Hd, Bn, sBH, sBH, sBS, scal, true, true, false);
    float* wsm = self_w_base + (long)l * Bn * Sq * Sq;
    softmax_q_kernel<<<(Bn * Sq) / 8, 256, 0, stream>>>(sc, wsm, wb, Sq);
    // x = x + w @ v
    launch_gemm(stream, wb, vb, nullptr, xc, xa, xba,
                Sq, Hd, Sq, Bn, sBS, sBH, sBH, 1.0f, false, false, false);
    { float* t = xc; xc = xa; xa = t; bf16* tb = xbc; xbc = xba; xba = tb; }

    // ---- encoder cross-attention ----
    launch_gemm(stream, xbc, Wq_eb + wOff, bq_e + bOff, nullptr, nullptr, qb,
                (int)MQ, Hd, Hd, 1, 0, 0, 0, 1.0f, false, false, false);
    launch_gemm(stream, annb, Wk_eb + wOff, bk_e + bOff, nullptr, nullptr, kb,
                (int)MQ, Hd, Hd, 1, 0, 0, 0, 1.0f, false, false, false);
    launch_gemm(stream, annb, Wv_eb + wOff, bv_e + bOff, nullptr, nullptr, vb,
                (int)MQ, Hd, Hd, 1, 0, 0, 0, 1.0f, false, false, false);
    launch_gemm(stream, qb, kb, nullptr, nullptr, sc, nullptr,
                Sq, Sq, Hd, Bn, sBH, sBH, sBS, scal, true, false, false);
    float* wem = enc_w_base + (long)l * Bn * Sq * Sq;
    softmax_q_kernel<<<(Bn * Sq) / 8, 256, 0, stream>>>(sc, wem, wb, Sq);
    launch_gemm(stream, wb, vb, nullptr, xc, xa, xba,
                Sq, Hd, Sq, Bn, sBS, sBH, sBH, 1.0f, false, false, false);
    { float* t = xc; xc = xa; xa = t; bf16* tb = xbc; xbc = xba; xba = tb; }

    // ---- MLP: x = x + relu(x @ Wm + bm) ----
    launch_gemm(stream, xbc, Wmb + wOff, bm + bOff, xc, xa, xba,
                (int)MQ, Hd, Hd, 1, 0, 0, 0, 1.0f, false, false, true);
    { float* t = xc; xc = xa; xa = t; bf16* tb = xbc; xbc = xba; xba = tb; }
  }

  // ---- output projection: [8192,1024] x [1024,10000] + bout ----
  launch_gemm(stream, xbc, Woutb, bout, nullptr, out_logits, nullptr,
              (int)MQ, Vc, Hd, 1, 0, 0, 0, 1.0f, false, false, false);
}
